// GraphTransformerClassifier_33346126086716
// MI455X (gfx1250) — matrix operand order repaired and checked by reading, not compiled, not run on hardware
//
#include <hip/hip_runtime.h>
#include <hip/hip_bf16.h>
#include <math.h>

#define N_NODES   100000
#define N_EDGES   1600000
#define F_IN      128
#define HID       64
#define NHEAD     4
#define HDIM      16
#define N_GRAPHS  512
#define N_CLASSES 10

typedef __attribute__((ext_vector_type(16))) _Float16 v16h;
typedef __attribute__((ext_vector_type(8)))  float    v8f;

// ---------------------------------------------------------------------------
// Operand packing: convert f32 activations/weights into WMMA-fragment-ordered
// f16 so the GEMM inner loop is pure 32B fragment loads + v_wmma.
//
// A fragment (16x32, 16-bit) lane layout per CDNA5 ISA:
//   lane rl (0..15) holds row rl; hsel = lane>>4 selects K-halves
//   VGPR v<4 -> K = hsel*8 + 2v, 2v+1 ; v>=4 -> K = 16 + hsel*8 + 2(v-4) ...
// B fragment (32x16): same K order as A (K = hsel*8 + j, then 16 + hsel*8 + j-8),
//   2 consecutive K per VGPR, column = lane&15.
// Packed layout: fragment index * 32 lanes * 16 halves, contiguous per lane.
// ---------------------------------------------------------------------------
__global__ void pack_a_kernel(const float* __restrict__ X, int Fin,
                              _Float16* __restrict__ Ap) {
  int t = blockIdx.x * blockDim.x + threadIdx.x;
  const int ksteps = Fin >> 5;
  const int total = (N_NODES / 16) * ksteps * 32;
  if (t >= total) return;
  int lane = t & 31;
  int s  = (t >> 5) % ksteps;
  int rt = (t >> 5) / ksteps;
  int rl = lane & 15, hsel = lane >> 4;
  const float* xr = X + (size_t)(rt * 16 + rl) * Fin + s * 32;
  _Float16* op = Ap + (size_t)t * 16;
#pragma unroll
  for (int vv = 0; vv < 8; ++vv) {
    int kb = (vv < 4) ? (hsel * 8 + 2 * vv) : (16 + hsel * 8 + 2 * (vv - 4));
    op[2 * vv]     = (_Float16)xr[kb];
    op[2 * vv + 1] = (_Float16)xr[kb + 1];
  }
}

struct PackWParams { const float* W[4]; };

__global__ void pack_w_kernel(PackWParams p, int Fin, _Float16* __restrict__ Wp) {
  int t = blockIdx.x * blockDim.x + threadIdx.x;
  const int ksteps = Fin >> 5;
  const int total = 4 * 4 * ksteps * 32;          // mat, ntile, kstep, lane
  if (t >= total) return;
  int lane = t & 31;
  int u = t >> 5;
  int s  = u % ksteps;
  int nt = (u / ksteps) & 3;
  int m  = (u / ksteps) >> 2;
  int rl = lane & 15, hsel = lane >> 4;
  const float* W = p.W[m];
  int col = nt * 16 + rl;
  int kbase = s * 32 + hsel * 8;
  _Float16* op = Wp + (size_t)t * 16;
#pragma unroll
  for (int j = 0; j < 16; ++j)
    op[j] = (_Float16)W[(size_t)(kbase + j + (j >= 8 ? 8 : 0)) * HID + col];
}

// ---------------------------------------------------------------------------
// Fused projection GEMM: O[mat] = X @ W[mat] + B[mat]   (mat = q,k,v,skip)
// One wave computes a 16x64 tile: per K-step, 1 A-frag + 4 B-frag loads
// (2x global_load_b128 each) feeding 4 v_wmma_f32_16x16x32_f16.
// ---------------------------------------------------------------------------
struct GemmParams {
  const _Float16* Apack;
  const _Float16* Wpack;
  int Fin;
  const float* B[4];
  float* O[4];
};

__global__ __launch_bounds__(32) void gemm_qkvs_wmma(GemmParams p) {
  const int lane = threadIdx.x & 31;
  const int rt = blockIdx.x;
  const int m  = blockIdx.y;
  const int ksteps = p.Fin >> 5;

  const v16h* __restrict__ Ap = (const v16h*)p.Apack;
  const v16h* __restrict__ Wp = (const v16h*)p.Wpack;
  const float* __restrict__ Bv = p.B[m];
  float* __restrict__ O        = p.O[m];

  v8f acc[4] = {};
  for (int s = 0; s < ksteps; ++s) {
    v16h a = Ap[(size_t)(rt * ksteps + s) * 32 + lane];
#pragma unroll
    for (int nt = 0; nt < 4; ++nt) {
      v16h bf = Wp[(size_t)(((m * 4 + nt) * ksteps) + s) * 32 + lane];
      acc[nt] = __builtin_amdgcn_wmma_f32_16x16x32_f16(
          false, a, false, bf, (short)0, acc[nt], false, false);
    }
  }

  const int rl = lane & 15, hsel = lane >> 4;
  const int rbase = rt * 16 + hsel * 8;
#pragma unroll
  for (int nt = 0; nt < 4; ++nt) {
    int col = nt * 16 + rl;
    float bias = Bv[col];
#pragma unroll
    for (int r = 0; r < 8; ++r)
      O[(size_t)(rbase + r) * HID + col] = acc[nt][r] + bias;
  }
}

// ---------------------------------------------------------------------------
// Float atomic-max via order-preserving uint key (handles negatives).
// ---------------------------------------------------------------------------
__device__ __forceinline__ unsigned fkey(float x) {
  unsigned u = __float_as_uint(x);
  return (u & 0x80000000u) ? ~u : (u | 0x80000000u);
}
__device__ __forceinline__ float funkey(unsigned kk) {
  unsigned u = (kk & 0x80000000u) ? (kk & 0x7FFFFFFFu) : ~kk;
  return __uint_as_float(u);
}

// Pass 1: alpha[e,h] = (q[dst,h]·k[src,h]) / 4 ; segment max over dst
__global__ void edge_alpha_kernel(const int* __restrict__ src, const int* __restrict__ dst,
                                  const float* __restrict__ q, const float* __restrict__ k,
                                  float* __restrict__ alpha, unsigned* __restrict__ amax) {
  int t = blockIdx.x * blockDim.x + threadIdx.x;
  if (t >= N_EDGES * NHEAD) return;
  int e = t >> 2, hh = t & 3;
  int s = src[e], d = dst[e];
  const float4* qp = (const float4*)(q + (size_t)d * HID + hh * HDIM);
  const float4* kp = (const float4*)(k + (size_t)s * HID + hh * HDIM);
  float acc = 0.f;
#pragma unroll
  for (int i = 0; i < 4; ++i) {
    float4 a4 = qp[i], b4 = kp[i];
    acc += a4.x * b4.x + a4.y * b4.y + a4.z * b4.z + a4.w * b4.w;
  }
  acc *= 0.25f;  // 1/sqrt(16)
  alpha[t] = acc;
  atomicMax(&amax[(size_t)d * NHEAD + hh], fkey(acc));
}

// Pass 2: ex = exp(alpha - amax[dst]) ; segment sum over dst
__global__ void edge_exp_kernel(const int* __restrict__ dst, const unsigned* __restrict__ amax,
                                float* __restrict__ alpha, float* __restrict__ denom) {
  int t = blockIdx.x * blockDim.x + threadIdx.x;
  if (t >= N_EDGES * NHEAD) return;
  int e = t >> 2, hh = t & 3;
  int d = dst[e];
  float ex = expf(alpha[t] - funkey(amax[(size_t)d * NHEAD + hh]));
  alpha[t] = ex;
  atomicAdd(denom + (size_t)d * NHEAD + hh, ex);
}

// Pass 3: agg[dst,h,:] += (ex/denom) * v[src,h,:]
__global__ void edge_agg_kernel(const int* __restrict__ src, const int* __restrict__ dst,
                                const float* __restrict__ ex, const float* __restrict__ denom,
                                const float* __restrict__ v, float* __restrict__ agg) {
  int t = blockIdx.x * blockDim.x + threadIdx.x;
  if (t >= N_EDGES * NHEAD) return;
  int e = t >> 2, hh = t & 3;
  int s = src[e], d = dst[e];
  float attn = ex[t] / (denom[(size_t)d * NHEAD + hh] + 1e-16f);
  const float* vp = v + (size_t)s * HID + hh * HDIM;
  float* ap = agg + (size_t)d * HID + hh * HDIM;
#pragma unroll
  for (int i = 0; i < HDIM; ++i) atomicAdd(ap + i, attn * vp[i]);
}

// BatchNorm (eval) + ReLU
__global__ void bn_relu_kernel(const float* __restrict__ in,
                               const float* __restrict__ g, const float* __restrict__ b,
                               const float* __restrict__ m, const float* __restrict__ vv,
                               float* __restrict__ out) {
  int idx = blockIdx.x * blockDim.x + threadIdx.x;
  if (idx >= N_NODES * HID) return;
  int f = idx & (HID - 1);
  float val = (in[idx] - m[f]) * rsqrtf(vv[f] + 1e-5f) * g[f] + b[f];
  out[idx] = fmaxf(val, 0.f);
}

// global_max_pool: post-ReLU values >= 0, so int-bit atomicMax == float max
__global__ void pool_max_kernel(const float* __restrict__ h, const int* __restrict__ batch,
                                float* __restrict__ gpool) {
  int idx = blockIdx.x * blockDim.x + threadIdx.x;
  if (idx >= N_NODES * HID) return;
  int n = idx >> 6, f = idx & 63;
  int gi = batch[n];
  atomicMax((int*)(gpool + (size_t)gi * HID + f), __float_as_int(h[idx]));
}

// MLP head: relu(g@p1+b) @ p2 + b2 -> relu(bn(·@cl1+b)) -> cl2 -> log_softmax
__global__ __launch_bounds__(64) void mlp_head(
    const float* __restrict__ gpool,
    const float* __restrict__ p1w, const float* __restrict__ p1b,
    const float* __restrict__ p2w, const float* __restrict__ p2b,
    const float* __restrict__ cl1w, const float* __restrict__ cl1b,
    const float* __restrict__ bng, const float* __restrict__ bnb,
    const float* __restrict__ bnm, const float* __restrict__ bnv,
    const float* __restrict__ cl2w, const float* __restrict__ cl2b,
    float* __restrict__ out) {
  __shared__ float s0[HID];
  __shared__ float s1[HID];
  __shared__ float lg[N_CLASSES];
  const int t = threadIdx.x;
  const int gi = blockIdx.x;

  s0[t] = gpool[(size_t)gi * HID + t];
  __syncthreads();

  float a = p1b[t];
  for (int j = 0; j < HID; ++j) a += s0[j] * p1w[j * HID + t];
  s1[t] = fmaxf(a, 0.f);
  __syncthreads();

  a = p2b[t];
  for (int j = 0; j < HID; ++j) a += s1[j] * p2w[j * HID + t];
  s0[t] = a;
  __syncthreads();

  a = cl1b[t];
  for (int j = 0; j < HID; ++j) a += s0[j] * cl1w[j * HID + t];
  a = (a - bnm[t]) * rsqrtf(bnv[t] + 1e-5f) * bng[t] + bnb[t];
  s1[t] = fmaxf(a, 0.f);
  __syncthreads();

  if (t < N_CLASSES) {
    float l = cl2b[t];
    for (int j = 0; j < HID; ++j) l += s1[j] * cl2w[j * N_CLASSES + t];
    lg[t] = l;
  }
  __syncthreads();

  if (t < N_CLASSES) {
    float mx = lg[0];
    for (int c = 1; c < N_CLASSES; ++c) mx = fmaxf(mx, lg[c]);
    float se = 0.f;
    for (int c = 0; c < N_CLASSES; ++c) se += expf(lg[c] - mx);
    out[(size_t)gi * N_CLASSES + t] = lg[t] - mx - logf(se);
  }
}

// ---------------------------------------------------------------------------
extern "C" void kernel_launch(void* const* d_in, const int* in_sizes, int n_in,
                              void* d_out, int out_size, void* d_ws, size_t ws_size,
                              hipStream_t stream) {
  (void)in_sizes; (void)n_in; (void)out_size; (void)ws_size;

  const float* x    = (const float*)d_in[0];
  const int*   ei   = (const int*)d_in[1];
  const int*   batch= (const int*)d_in[2];
  const int*   src  = ei;
  const int*   dst  = ei + N_EDGES;

  const float* c1w[4] = {(const float*)d_in[3], (const float*)d_in[5],
                         (const float*)d_in[7], (const float*)d_in[9]};
  const float* c1b[4] = {(const float*)d_in[4], (const float*)d_in[6],
                         (const float*)d_in[8], (const float*)d_in[10]};
  const float* c2w[4] = {(const float*)d_in[11], (const float*)d_in[13],
                         (const float*)d_in[15], (const float*)d_in[17]};
  const float* c2b[4] = {(const float*)d_in[12], (const float*)d_in[14],
                         (const float*)d_in[16], (const float*)d_in[18]};
  const float* bn1[4]  = {(const float*)d_in[19], (const float*)d_in[20],
                          (const float*)d_in[21], (const float*)d_in[22]};
  const float* bn2[4]  = {(const float*)d_in[23], (const float*)d_in[24],
                          (const float*)d_in[25], (const float*)d_in[26]};
  const float* clbn[4] = {(const float*)d_in[27], (const float*)d_in[28],
                          (const float*)d_in[29], (const float*)d_in[30]};
  const float* p1w = (const float*)d_in[31];
  const float* p2w = (const float*)d_in[32];
  const float* cl1w = (const float*)d_in[33];
  const float* cl2w = (const float*)d_in[34];
  const float* p1b = (const float*)d_in[35];
  const float* p2b = (const float*)d_in[36];
  const float* cl1b = (const float*)d_in[37];
  const float* cl2b = (const float*)d_in[38];
  float* out = (float*)d_out;

  // workspace layout (floats). The A-pack f16 buffer time-shares the alpha
  // region: A-pack is live only [pack -> gemm], alpha only [edge passes].
  // Layer-1 A-pack = N*128 halves == E*NHEAD floats exactly.
  const size_t NF = (size_t)N_NODES * HID;            // 6.4M floats
  const size_t NH = (size_t)N_NODES * NHEAD;          // 0.4M floats
  float* ws    = (float*)d_ws;
  float* q     = ws;
  float* k     = ws + NF;
  float* v     = ws + 2 * NF;
  float* agg   = ws + 3 * NF;                          // init'd by skip GEMM
  float* h     = ws + 4 * NF;
  float* alpha = ws + 5 * NF;                          // E*NHEAD floats
  _Float16* apack = (_Float16*)alpha;                  // time-shared
  unsigned* amax = (unsigned*)(ws + 6 * NF);
  float* denom = ws + 6 * NF + NH;
  float* gpool = ws + 6 * NF + 2 * NH;
  _Float16* wpack = (_Float16*)(gpool + (size_t)N_GRAPHS * HID);  // 32K halves max

  const dim3 gemmBlk(32);
  const dim3 gemmGrid(N_NODES / 16, 4);
  const int  ETH = N_EDGES * NHEAD;
  const int  edgeBlocks = (ETH + 255) / 256;
  const int  nodeBlocks = (N_NODES * HID + 255) / 256;
  const int  packA1 = (N_NODES / 16) * (F_IN / 32) * 32;   // 800000
  const int  packA2 = (N_NODES / 16) * (HID  / 32) * 32;   // 400000
  const int  packW1 = 4 * 4 * (F_IN / 32) * 32;            // 2048
  const int  packW2 = 4 * 4 * (HID  / 32) * 32;            // 1024

  // ---- Layer 1 ----
  hipMemsetAsync(amax,  0, NH * sizeof(unsigned), stream);
  hipMemsetAsync(denom, 0, NH * sizeof(float), stream);
  hipMemsetAsync(gpool, 0, (size_t)N_GRAPHS * HID * sizeof(float), stream);

  PackWParams pw1; for (int i = 0; i < 4; ++i) pw1.W[i] = c1w[i];
  pack_w_kernel<<<(packW1 + 255) / 256, 256, 0, stream>>>(pw1, F_IN, wpack);
  pack_a_kernel<<<(packA1 + 255) / 256, 256, 0, stream>>>(x, F_IN, apack);

  GemmParams g1;
  g1.Apack = apack; g1.Wpack = wpack; g1.Fin = F_IN;
  for (int i = 0; i < 4; ++i) g1.B[i] = c1b[i];
  g1.O[0] = q; g1.O[1] = k; g1.O[2] = v; g1.O[3] = agg;
  gemm_qkvs_wmma<<<gemmGrid, gemmBlk, 0, stream>>>(g1);

  edge_alpha_kernel<<<edgeBlocks, 256, 0, stream>>>(src, dst, q, k, alpha, amax);
  edge_exp_kernel<<<edgeBlocks, 256, 0, stream>>>(dst, amax, alpha, denom);
  edge_agg_kernel<<<edgeBlocks, 256, 0, stream>>>(src, dst, alpha, denom, v, agg);
  bn_relu_kernel<<<nodeBlocks, 256, 0, stream>>>(agg, bn1[0], bn1[1], bn1[2], bn1[3], h);

  // ---- Layer 2 ----
  hipMemsetAsync(amax,  0, NH * sizeof(unsigned), stream);
  hipMemsetAsync(denom, 0, NH * sizeof(float), stream);

  PackWParams pw2; for (int i = 0; i < 4; ++i) pw2.W[i] = c2w[i];
  pack_w_kernel<<<(packW2 + 255) / 256, 256, 0, stream>>>(pw2, HID, wpack);
  pack_a_kernel<<<(packA2 + 255) / 256, 256, 0, stream>>>(h, HID, apack);

  GemmParams g2;
  g2.Apack = apack; g2.Wpack = wpack; g2.Fin = HID;
  for (int i = 0; i < 4; ++i) g2.B[i] = c2b[i];
  g2.O[0] = q; g2.O[1] = k; g2.O[2] = v; g2.O[3] = agg;
  gemm_qkvs_wmma<<<gemmGrid, gemmBlk, 0, stream>>>(g2);

  edge_alpha_kernel<<<edgeBlocks, 256, 0, stream>>>(src, dst, q, k, alpha, amax);
  edge_exp_kernel<<<edgeBlocks, 256, 0, stream>>>(dst, amax, alpha, denom);
  edge_agg_kernel<<<edgeBlocks, 256, 0, stream>>>(src, dst, alpha, denom, v, agg);
  bn_relu_kernel<<<nodeBlocks, 256, 0, stream>>>(agg, bn2[0], bn2[1], bn2[2], bn2[3], h);

  // ---- Pool + head ----
  pool_max_kernel<<<nodeBlocks, 256, 0, stream>>>(h, batch, gpool);
  mlp_head<<<N_GRAPHS, 64, 0, stream>>>(gpool, p1w, p1b, p2w, p2b,
                                        cl1w, cl1b, clbn[0], clbn[1], clbn[2], clbn[3],
                                        cl2w, cl2b, out);
}
